// SemanticProjector_82635170775492
// MI455X (gfx1250) — matrix operand order repaired and checked by reading, not compiled, not run on hardware
//
#include <hip/hip_runtime.h>
#include <hip/hip_bf16.h>

// ---------------------------------------------------------------------------
// SemanticProjector for MI455X (gfx1250, wave32, WMMA)
//   B=32, N=4096, C=768, K=64
//   logits = x @ W^T  -> softmax over N -> tokens = attn^T @ x
// Memory-bound (x = 402MB fp32 read twice => ~35us floor @ 23.3TB/s).
// Strategy: bf16 WMMA (f32 accum), exp(logits) stored transposed in bf16
// workspace, 1/sum(exp) folded into pass-2 epilogue.
// Round 3: v_perm-based f32->bf16 packing (3 VALU/pair), b128 LDS reads for
// pass-2 A fragments, async global->LDS Et staging (ASYNCcnt path).
// ---------------------------------------------------------------------------

#define Bsz 32
#define Nn  4096
#define Cc  768
#define Kk  64

typedef __attribute__((ext_vector_type(16))) __bf16 v16bf;
typedef __attribute__((ext_vector_type(8)))  float  v8f;

union bf16x16 { unsigned u[8]; v16bf v; };

__device__ inline unsigned bf16_rne_sw(float a) {
    union { float f; unsigned u; } x; x.f = a;
    return (x.u + 0x7FFFu + ((x.u >> 16) & 1u)) >> 16;
}

#if __has_builtin(__builtin_amdgcn_cvt_pk_bf16_f32)
typedef __attribute__((ext_vector_type(2))) __bf16 v2bf;
__device__ inline unsigned pk_bf16(float a, float b) {   // a -> [15:0], b -> [31:16]
    union { v2bf v; unsigned u; } c;
    c.v = __builtin_amdgcn_cvt_pk_bf16_f32(a, b);
    return c.u;
}
#elif __has_builtin(__builtin_amdgcn_perm)
// Round-half-up to bf16 then pack both high halves with one v_perm_b32.
__device__ inline unsigned pk_bf16(float a, float b) {
    union { float f; unsigned u; } ua, ub; ua.f = a; ub.f = b;
    const unsigned ra = ua.u + 0x8000u;
    const unsigned rb = ub.u + 0x8000u;
    // sel 0x07060302: dst.b0=ra.b2 dst.b1=ra.b3 dst.b2=rb.b2 dst.b3=rb.b3
    return __builtin_amdgcn_perm(rb, ra, 0x07060302u);
}
#else
__device__ inline unsigned pk_bf16(float a, float b) {
    return (bf16_rne_sw(a) & 0xFFFFu) | (bf16_rne_sw(b) << 16);
}
#endif

// ---------------------------------------------------------------------------
// Pass 1: logits tile (16n x 32k per wave) via v_wmma_f32_16x16x32_bf16,
// exp + per-(b,k) sum-of-exp atomics, exp stored transposed as Et[b][k][n] bf16.
// Grid: 32 n-tiles * 2 k-halves * 32 batches = 2048 blocks of 256 threads.
// LDS: 48KB W B-fragments + 12KB per-wave transpose = 60KB.
// ---------------------------------------------------------------------------
__global__ __launch_bounds__(256)
void sp_pass1(const float* __restrict__ x, const float* __restrict__ W,
              unsigned short* __restrict__ Et, float* __restrict__ sumexp) {
    // [chunk(24)][ktl(2)][lane(32)][j(8)] u32 : exact WMMA B-fragment layout
    __shared__ unsigned wfrag[24 * 2 * 32 * 8];          // 49152 B
    __shared__ unsigned short etr[8][32 * 24];           // per-wave 32 rows * 48B (padded)

    const int tid = threadIdx.x;
    const int blk = blockIdx.x;
    const int nt = blk & 31;            // n tile group (128 n per wg)
    const int kh = (blk >> 5) & 1;      // k half (32 k per wg)
    const int b  = blk >> 6;            // batch

    // Build bf16 B-fragments of W in LDS (each W element stored once: 48KB).
    for (int u = tid; u < 24 * 2 * 32 * 8; u += 256) {
        const int j     = u & 7;
        const int lane  = (u >> 3) & 31;
        const int ktl   = (u >> 8) & 1;
        const int chunk = u >> 9;
        const int k = kh * 32 + ktl * 16 + (lane & 15);          // B column = lane%16
        const int c = chunk * 32 + ((j & 4) << 2) + ((lane >> 4) << 3) + 2 * (j & 3);   // B K-dim = (j<4?0:16)+8*half+2*(j&3)
        const float* wp = W + k * Cc + c;
        wfrag[u] = pk_bf16(wp[0], wp[1]);
    }
    __syncthreads();

    const int w    = tid >> 5;
    const int lane = tid & 31;
    const int m    = lane & 15;
    const int half = lane >> 4;
    const int n0   = nt * 128 + w * 16;

    const float* xrow = x + ((long)b * Nn + (n0 + m)) * Cc;

    v8f acc0 = {}; v8f acc1 = {};

    for (int chunk = 0; chunk < 24; ++chunk) {
        const int c0 = chunk * 32;
        // A fragment: lane holds x[row=n0+m][c0 + kiA(j)], kiA=(j<4?0:16)+8*half+2*(j&3)
        const float4* p0 = (const float4*)(xrow + c0 + 8 * half);
        const float4  a0 = p0[0], a1 = p0[1];
        const float4* p1 = (const float4*)(xrow + c0 + 16 + 8 * half);
        const float4  a2 = p1[0], a3 = p1[1];
        bf16x16 A;
        A.u[0] = pk_bf16(a0.x, a0.y); A.u[1] = pk_bf16(a0.z, a0.w);
        A.u[2] = pk_bf16(a1.x, a1.y); A.u[3] = pk_bf16(a1.z, a1.w);
        A.u[4] = pk_bf16(a2.x, a2.y); A.u[5] = pk_bf16(a2.z, a2.w);
        A.u[6] = pk_bf16(a3.x, a3.y); A.u[7] = pk_bf16(a3.z, a3.w);

        {   // k tile 0
            bf16x16 Bm;
            const uint4* q = (const uint4*)&wfrag[((chunk * 2 + 0) * 32 + lane) * 8];
            const uint4 b0 = q[0], b1 = q[1];
            Bm.u[0] = b0.x; Bm.u[1] = b0.y; Bm.u[2] = b0.z; Bm.u[3] = b0.w;
            Bm.u[4] = b1.x; Bm.u[5] = b1.y; Bm.u[6] = b1.z; Bm.u[7] = b1.w;
            acc0 = __builtin_amdgcn_wmma_f32_16x16x32_bf16(false, A.v, false, Bm.v,
                                                           (short)0, acc0, false, false);
        }
        {   // k tile 1
            bf16x16 Bm;
            const uint4* q = (const uint4*)&wfrag[((chunk * 2 + 1) * 32 + lane) * 8];
            const uint4 b0 = q[0], b1 = q[1];
            Bm.u[0] = b0.x; Bm.u[1] = b0.y; Bm.u[2] = b0.z; Bm.u[3] = b0.w;
            Bm.u[4] = b1.x; Bm.u[5] = b1.y; Bm.u[6] = b1.z; Bm.u[7] = b1.w;
            acc1 = __builtin_amdgcn_wmma_f32_16x16x32_bf16(false, A.v, false, Bm.v,
                                                           (short)0, acc1, false, false);
        }
    }

    // exp + partial softmax denominators + per-wave LDS transpose.
    // C/D layout: acc[j] lane L -> n = n0 + j + 8*half, k-col = lane%16.
    unsigned short* tr = etr[w];
    float ps0 = 0.f, ps1 = 0.f;
#pragma unroll
    for (int j = 0; j < 8; ++j) {
        const float e0 = __expf(acc0[j]);
        const float e1 = __expf(acc1[j]);
        ps0 += e0; ps1 += e1;
        const int nl = j + 8 * half;
        const unsigned ue = pk_bf16(e0, e1);
        tr[(m)      * 24 + nl] = (unsigned short)(ue & 0xFFFFu);
        tr[(16 + m) * 24 + nl] = (unsigned short)(ue >> 16);
    }
    atomicAdd(&sumexp[b * Kk + kh * 32 + m],      ps0);
    atomicAdd(&sumexp[b * Kk + kh * 32 + 16 + m], ps1);

    // Coalesced transposed store: lane owns row k_local = lane (32 rows of 16 n).
    const uint4 r0 = *(const uint4*)&tr[lane * 24];
    const uint4 r1 = *(const uint4*)&tr[lane * 24 + 8];
    unsigned short* dst = Et + ((long)(b * Kk + kh * 32 + lane) * Nn + n0);
    ((uint4*)dst)[0] = r0;
    ((uint4*)dst)[1] = r1;
}

// ---------------------------------------------------------------------------
// Pass 2: tokens[b][k][c] = (1/sumexp[b][k]) * sum_n Et[b][k][n] * x[b][n][c]
// Grid: 32 batches * 12 c-groups (64 cols) = 384 blocks of 128 threads.
// Et chunk (64k x 128n) staged in LDS via async global->LDS copies (ASYNCcnt);
// x read exactly once (B operand).
// ---------------------------------------------------------------------------
__global__ __launch_bounds__(128)
void sp_pass2(const float* __restrict__ x, const unsigned short* __restrict__ Et,
              const float* __restrict__ sumexp, float* __restrict__ out) {
    __shared__ unsigned els[64 * 68];   // [k][np] bf16-pairs, padded stride 68 u32

    const int tid = threadIdx.x;
    const int blk = blockIdx.x;
    const int cg  = blk % 12;
    const int b   = blk / 12;
    const int w    = tid >> 5;
    const int lane = tid & 31;
    const int m    = lane & 15;
    const int half = lane >> 4;
    const int c0   = cg * 64 + w * 16;

    v8f acc[4] = {v8f{}, v8f{}, v8f{}, v8f{}};

    const float* xb = x + (long)b * Nn * Cc;
    const unsigned short* eb = Et + (long)b * Kk * Nn;

    // Per-thread async-copy source/dest (thread = (k-row, 128B half-row)).
    const int fk  = tid >> 1;
    const int fh2 = tid & 1;
    const unsigned long long gsrc0 =
        (unsigned long long)(size_t)(eb + (long)fk * Nn + fh2 * 64);
    // Low 32 bits of a generic pointer to LDS are the LDS byte offset.
    const unsigned ldst0 = (unsigned)(size_t)(void*)&els[fk * 68 + fh2 * 32];

    for (int no = 0; no < 32; ++no) {               // 128 n per outer chunk
        {   // async fill: memory -> LDS directly, no VGPR round-trip
            const unsigned long long g = gsrc0 + (unsigned long long)no * 256;
#pragma unroll
            for (int i = 0; i < 8; ++i) {
                const unsigned long long ga = g + (unsigned)(i * 16);
                const unsigned la = ldst0 + (unsigned)(i * 16);
                asm volatile("global_load_async_to_lds_b128 %0, %1, off"
                             :: "v"(la), "v"(ga) : "memory");
            }
            asm volatile("s_wait_asynccnt 0x0" ::: "memory");
        }
        __syncthreads();

#pragma unroll
        for (int sub = 0; sub < 4; ++sub) {         // 32-n WMMA steps
            const int nb = no * 128 + sub * 32;
            // B fragment = x[n][c]: K-dim n = nb+(j<4?0:16)+8*half+2*(j&3)(+1), col c = c0+m
            bf16x16 Bm;
#pragma unroll
            for (int j = 0; j < 8; ++j) {
                const int n = nb + 8 * half + 2 * (j & 3) + ((j & 4) << 2);
                const float* xp = xb + (long)n * Cc + c0 + m;
                Bm.u[j] = pk_bf16(xp[0], xp[Cc]);
            }
#pragma unroll
            for (int kt = 0; kt < 4; ++kt) {
                // A fragment from LDS: row k = kt*16+m; j=0..3 and j=4..7 are
                // contiguous np-runs -> two ds_load_b128 per k-tile.
                const unsigned* arow = &els[(kt * 16 + m) * 68 + sub * 16 + 4 * half];
                const uint4 q0 = *(const uint4*)&arow[0];
                const uint4 q1 = *(const uint4*)&arow[8];
                bf16x16 A;
                A.u[0] = q0.x; A.u[1] = q0.y; A.u[2] = q0.z; A.u[3] = q0.w;
                A.u[4] = q1.x; A.u[5] = q1.y; A.u[6] = q1.z; A.u[7] = q1.w;
                acc[kt] = __builtin_amdgcn_wmma_f32_16x16x32_bf16(false, A.v, false, Bm.v,
                                                                  (short)0, acc[kt], false, false);
            }
        }
        __syncthreads();
    }

    // Epilogue: scale rows by 1/sumexp and store (every (b,k,c) written once).
#pragma unroll
    for (int kt = 0; kt < 4; ++kt) {
#pragma unroll
        for (int j = 0; j < 8; ++j) {
            const int k = kt * 16 + j + 8 * half;       // C/D: M = j + 8*half
            const float s = 1.0f / sumexp[b * Kk + k];
            out[(long)(b * Kk + k) * Cc + c0 + m] = acc[kt][j] * s;
        }
    }
}

// ---------------------------------------------------------------------------
extern "C" void kernel_launch(void* const* d_in, const int* in_sizes, int n_in,
                              void* d_out, int out_size, void* d_ws, size_t ws_size,
                              hipStream_t stream) {
    (void)in_sizes; (void)n_in; (void)out_size; (void)ws_size;
    const float* x = (const float*)d_in[0];          // [32,4096,768] f32
    const float* W = (const float*)d_in[1];          // [64,768] f32
    float* out = (float*)d_out;                      // [32,64,768] f32

    // Workspace: Et bf16 [32][64][4096] (16.78MB) + sumexp f32 [32][64]
    unsigned short* Et = (unsigned short*)d_ws;
    float* sumexp = (float*)((char*)d_ws + (size_t)Bsz * Kk * Nn * 2);

    hipMemsetAsync(sumexp, 0, (size_t)Bsz * Kk * sizeof(float), stream);
    sp_pass1<<<dim3(Bsz * 2 * 32), dim3(256), 0, stream>>>(x, W, Et, sumexp);
    sp_pass2<<<dim3(Bsz * 12), dim3(128), 0, stream>>>(x, Et, sumexp, out);
}
